// SparseAttention_69303592288297
// MI455X (gfx1250) — matrix operand order repaired and checked by reading, not compiled, not run on hardware
//
#include <hip/hip_runtime.h>

typedef __bf16 bf16_t;
typedef __attribute__((ext_vector_type(16))) __bf16 v16bf;
typedef __attribute__((ext_vector_type(8)))  __bf16 v8bf;
typedef __attribute__((ext_vector_type(8)))  float  v8f;

#define SHUF16 0,1,2,3,4,5,6,7,8,9,10,11,12,13,14,15

// ---------------------------------------------------------------------------
// Fragment loaders matching the CDNA5 WMMA VGPR layouts (CDNA5 ISA)
// ---------------------------------------------------------------------------

// A-matrix 16x32 (MxK), 16-bit. Lane L (<16): row L, K = {0..7, 16..23};
// lane L+16: row L, K = {8..15, 24..31}. Source must be row-major, K-contiguous.
__device__ __forceinline__ v16bf load_a_16x32(const bf16_t* base, int ld, int row0, int k0) {
  const int lane = threadIdx.x & 31;
  const int r = lane & 15;
  const int h = lane >> 4;
  const bf16_t* p = base + (size_t)(row0 + r) * (size_t)ld + (size_t)(k0 + h * 8);
  v8bf lo = *reinterpret_cast<const v8bf*>(p);
  v8bf hi = *reinterpret_cast<const v8bf*>(p + 16);
  return __builtin_shufflevector(lo, hi, SHUF16);
}

// B-matrix 32x16 (KxN), 16-bit. Lane n (<16): column n, K = {0..7, 16..23};
// lane n+16: column n, K = {8..15, 24..31}. Source stored with each column K-contiguous
// (i.e. memory layout [N][K] row-major, leading dim = ld).
__device__ __forceinline__ v16bf load_b_32x16(const bf16_t* base, int ld, int n0, int k0) {
  const int lane = threadIdx.x & 31;
  const int n = lane & 15;
  const int h = lane >> 4;
  const bf16_t* p = base + (size_t)(n0 + n) * (size_t)ld + (size_t)(k0 + h * 8);
  v8bf lo = *reinterpret_cast<const v8bf*>(p);
  v8bf hi = *reinterpret_cast<const v8bf*>(p + 16);
  return __builtin_shufflevector(lo, hi, SHUF16);
}

__device__ __forceinline__ v8f wmma_bf16(v16bf a, v16bf b, v8f c) {
  return __builtin_amdgcn_wmma_f32_16x16x32_bf16(false, a, false, b, (short)0, c, false, false);
}

// ---------------------------------------------------------------------------
// fp32 -> bf16 conversion
// ---------------------------------------------------------------------------
__global__ void f32_to_bf16(const float* __restrict__ in, bf16_t* __restrict__ out, int n) {
  int i = blockIdx.x * blockDim.x + threadIdx.x;
  if (i < n) out[i] = (bf16_t)in[i];
}

// ---------------------------------------------------------------------------
// QKV projection: y = x @ W^T   (x: [4096][1024] bf16, W: [1024][1024] bf16)
// One wave computes a 32x64 output tile (B fragments reused across 2 A tiles).
// which = 0:Q 1:K 2:V(transposed out)
// ---------------------------------------------------------------------------
__global__ void qkv_gemm(const bf16_t* __restrict__ xb,
                         const bf16_t* __restrict__ wqb,
                         const bf16_t* __restrict__ wkb,
                         const bf16_t* __restrict__ wvb,
                         bf16_t* __restrict__ Qb,
                         bf16_t* __restrict__ Kb,
                         bf16_t* __restrict__ VTb) {
  const int r0 = blockIdx.x * 32;
  const int n0 = blockIdx.y * 64;
  const int which = blockIdx.z;
  const bf16_t* W = (which == 0) ? wqb : (which == 1) ? wkb : wvb;

  const v8f zero8 = {0.f, 0.f, 0.f, 0.f, 0.f, 0.f, 0.f, 0.f};
  v8f acc[2][4] = {{zero8, zero8, zero8, zero8}, {zero8, zero8, zero8, zero8}};

  for (int k0 = 0; k0 < 1024; k0 += 32) {
    v16bf a0 = load_a_16x32(xb, 1024, r0, k0);
    v16bf a1 = load_a_16x32(xb, 1024, r0 + 16, k0);
#pragma unroll
    for (int g = 0; g < 4; ++g) {
      v16bf b = load_b_32x16(W, 1024, n0 + g * 16, k0);
      acc[0][g] = wmma_bf16(a0, b, acc[0][g]);
      acc[1][g] = wmma_bf16(a1, b, acc[1][g]);
    }
  }

  const int lane = threadIdx.x & 31;
  const int nl = lane & 15;
  const int hi8 = (lane >> 4) * 8;   // C/D layout: row = vgpr + 8*(lane>=16)

  if (which == 2) {
    // Store V transposed: VT[feat][token], feat-major with token contiguous
#pragma unroll
    for (int mi = 0; mi < 2; ++mi)
#pragma unroll
      for (int g = 0; g < 4; ++g)
#pragma unroll
        for (int v = 0; v < 8; ++v)
          VTb[(size_t)(n0 + g * 16 + nl) * 4096 + (size_t)(r0 + mi * 16 + v + hi8)] =
              (bf16_t)acc[mi][g][v];
  } else {
    bf16_t* dst = (which == 0) ? Qb : Kb;
#pragma unroll
    for (int mi = 0; mi < 2; ++mi)
#pragma unroll
      for (int g = 0; g < 4; ++g)
#pragma unroll
        for (int v = 0; v < 8; ++v)
          dst[(size_t)(r0 + mi * 16 + v + hi8) * 1024 + (size_t)(n0 + g * 16 + nl)] =
              (bf16_t)acc[mi][g][v];
  }
}

// ---------------------------------------------------------------------------
// Sparse flash attention.  One wave per (16-query block, head).
// Window: key tiles [r0-256, r0+15]; strided: diagonal tiles c0 = r0-64j, j>=5.
// Exact element mask: d=r-c valid iff d>=0 && (d<256 || d%64==0).
// ---------------------------------------------------------------------------
__global__ void sparse_attn(const bf16_t* __restrict__ Qb,
                            const bf16_t* __restrict__ Kb,
                            const bf16_t* __restrict__ VTb,
                            bf16_t* __restrict__ Ab) {
  __shared__ __align__(16) bf16_t plds[16 * 32];

  const int r0 = blockIdx.x * 16;
  const int head = blockIdx.y;
  const bf16_t* qh = Qb + head * 64;
  const bf16_t* kh = Kb + head * 64;
  const bf16_t* vth = VTb + (size_t)head * 64 * 4096;

  const int lane = threadIdx.x & 31;
  const int nl = lane & 15;
  const int hi8 = (lane >> 4) * 8;

  // Q fragments (K-dim = head_dim 64 -> two K=32 chunks)
  v16bf aq0 = load_a_16x32(qh, 1024, r0, 0);
  v16bf aq1 = load_a_16x32(qh, 1024, r0, 32);

  const v8f zero8 = {0.f, 0.f, 0.f, 0.f, 0.f, 0.f, 0.f, 0.f};
  const float SCALE = 0.125f * 1.44269504088896f;  // head_dim^-0.5 * log2(e)
  const float NEG = -3.0e38f;

  v8f m, l = zero8, o0 = zero8, o1 = zero8, o2 = zero8, o3 = zero8;
#pragma unroll
  for (int v = 0; v < 8; ++v) m[v] = NEG;

  auto doblk = [&](int c0) {
    // ---- scores: 16 queries x 32 keys (two 16x16 WMMA tiles) ----
    v8f s0 = zero8, s1 = zero8;
    v16bf b0 = load_b_32x16(kh, 1024, c0, 0);
    v16bf b1 = load_b_32x16(kh, 1024, c0, 32);
    s0 = wmma_bf16(aq0, b0, s0);
    s0 = wmma_bf16(aq1, b1, s0);
    b0 = load_b_32x16(kh, 1024, c0 + 16, 0);
    b1 = load_b_32x16(kh, 1024, c0 + 16, 32);
    s1 = wmma_bf16(aq0, b0, s1);
    s1 = wmma_bf16(aq1, b1, s1);

    // ---- mask + scale (log2-domain) ----
    v8f bm;
#pragma unroll
    for (int v = 0; v < 8; ++v) {
      int r = r0 + v + hi8;
      int d0 = r - (c0 + nl);
      int d1 = d0 - 16;
      bool ok0 = (d0 >= 0) && ((d0 < 256) || ((d0 & 63) == 0));
      bool ok1 = (d1 >= 0) && ((d1 < 256) || ((d1 & 63) == 0));
      s0[v] = ok0 ? s0[v] * SCALE : NEG;
      s1[v] = ok1 ? s1[v] * SCALE : NEG;
      bm[v] = fmaxf(s0[v], s1[v]);
    }
    // row max across the 16 lanes holding one score row
#pragma unroll
    for (int v = 0; v < 8; ++v) {
      float x = bm[v];
      x = fmaxf(x, __shfl_xor(x, 1, 16));
      x = fmaxf(x, __shfl_xor(x, 2, 16));
      x = fmaxf(x, __shfl_xor(x, 4, 16));
      x = fmaxf(x, __shfl_xor(x, 8, 16));
      bm[v] = x;
    }
    // ---- online softmax update ----
    v8f p0, p1, rs;
#pragma unroll
    for (int v = 0; v < 8; ++v) {
      float mn = fmaxf(m[v], bm[v]);
      float al = __builtin_amdgcn_exp2f(m[v] - mn);
      m[v] = mn;
      p0[v] = __builtin_amdgcn_exp2f(s0[v] - mn);
      p1[v] = __builtin_amdgcn_exp2f(s1[v] - mn);
      rs[v] = p0[v] + p1[v];
      l[v] *= al;
      o0[v] *= al; o1[v] *= al; o2[v] *= al; o3[v] *= al;
    }
#pragma unroll
    for (int v = 0; v < 8; ++v) {
      float x = rs[v];
      x += __shfl_xor(x, 1, 16);
      x += __shfl_xor(x, 2, 16);
      x += __shfl_xor(x, 4, 16);
      x += __shfl_xor(x, 8, 16);
      l[v] += x;
    }
    // ---- C-layout -> A-layout via LDS, then P @ V ----
    __syncthreads();
#pragma unroll
    for (int v = 0; v < 8; ++v) {
      int row = v + hi8;
      plds[row * 32 + nl]      = (bf16_t)p0[v];
      plds[row * 32 + 16 + nl] = (bf16_t)p1[v];
    }
    __syncthreads();
    v16bf pa = load_a_16x32(plds, 32, 0, 0);
    o0 = wmma_bf16(pa, load_b_32x16(vth, 4096, 0,  c0), o0);
    o1 = wmma_bf16(pa, load_b_32x16(vth, 4096, 16, c0), o1);
    o2 = wmma_bf16(pa, load_b_32x16(vth, 4096, 32, c0), o2);
    o3 = wmma_bf16(pa, load_b_32x16(vth, 4096, 48, c0), o3);
  };

  // window (+ d==256 stride hit) tiles
  int w0 = (r0 >= 256) ? (r0 - 256) : 0;
  for (int c0 = w0; c0 <= r0 + 15; c0 += 32) doblk(c0);
  // remaining strided hits: diagonal tiles, j >= 5
  for (int c0 = r0 - 320; c0 >= 0; c0 -= 64) doblk(c0);

  // ---- normalize and store (heads concatenated -> [T][1024] bf16) ----
#pragma unroll
  for (int v = 0; v < 8; ++v) {
    float inv = 1.0f / l[v];
    size_t off = (size_t)(r0 + v + hi8) * 1024 + (size_t)(head * 64 + nl);
    Ab[off +  0] = (bf16_t)(o0[v] * inv);
    Ab[off + 16] = (bf16_t)(o1[v] * inv);
    Ab[off + 32] = (bf16_t)(o2[v] * inv);
    Ab[off + 48] = (bf16_t)(o3[v] * inv);
  }
}

// ---------------------------------------------------------------------------
// Output projection: out = attn @ Wo^T, fp32 stores to d_out.
// One wave computes a 32x64 output tile.
// ---------------------------------------------------------------------------
__global__ void out_gemm(const bf16_t* __restrict__ Ab,
                         const bf16_t* __restrict__ wob,
                         float* __restrict__ out) {
  const int r0 = blockIdx.x * 32;
  const int n0 = blockIdx.y * 64;

  const v8f zero8 = {0.f, 0.f, 0.f, 0.f, 0.f, 0.f, 0.f, 0.f};
  v8f acc[2][4] = {{zero8, zero8, zero8, zero8}, {zero8, zero8, zero8, zero8}};

  for (int k0 = 0; k0 < 1024; k0 += 32) {
    v16bf a0 = load_a_16x32(Ab, 1024, r0, k0);
    v16bf a1 = load_a_16x32(Ab, 1024, r0 + 16, k0);
#pragma unroll
    for (int g = 0; g < 4; ++g) {
      v16bf b = load_b_32x16(wob, 1024, n0 + g * 16, k0);
      acc[0][g] = wmma_bf16(a0, b, acc[0][g]);
      acc[1][g] = wmma_bf16(a1, b, acc[1][g]);
    }
  }

  const int lane = threadIdx.x & 31;
  const int nl = lane & 15;
  const int hi8 = (lane >> 4) * 8;
#pragma unroll
  for (int mi = 0; mi < 2; ++mi)
#pragma unroll
    for (int g = 0; g < 4; ++g)
#pragma unroll
      for (int v = 0; v < 8; ++v)
        out[(size_t)(r0 + mi * 16 + v + hi8) * 1024 + (size_t)(n0 + g * 16 + nl)] =
            acc[mi][g][v];
}

// ---------------------------------------------------------------------------
extern "C" void kernel_launch(void* const* d_in, const int* in_sizes, int n_in,
                              void* d_out, int out_size, void* d_ws, size_t ws_size,
                              hipStream_t stream) {
  (void)in_sizes; (void)n_in; (void)out_size; (void)ws_size;

  const float* x  = (const float*)d_in[0];
  const float* Wq = (const float*)d_in[1];
  const float* Wk = (const float*)d_in[2];
  const float* Wv = (const float*)d_in[3];
  const float* Wo = (const float*)d_in[4];

  char* ws = (char*)d_ws;
  const size_t MB = (size_t)1 << 20;
  bf16_t* xb  = (bf16_t*)(ws + 0 * MB);   // 8 MiB  [4096][1024]
  bf16_t* wqb = (bf16_t*)(ws + 8 * MB);   // 2 MiB
  bf16_t* wkb = (bf16_t*)(ws + 10 * MB);  // 2 MiB
  bf16_t* wvb = (bf16_t*)(ws + 12 * MB);  // 2 MiB
  bf16_t* wob = (bf16_t*)(ws + 14 * MB);  // 2 MiB
  bf16_t* Qb  = (bf16_t*)(ws + 16 * MB);  // 8 MiB  [4096][1024]
  bf16_t* Kb  = (bf16_t*)(ws + 24 * MB);  // 8 MiB  [4096][1024]
  bf16_t* VTb = (bf16_t*)(ws + 32 * MB);  // 8 MiB  [1024][4096] (V transposed)
  bf16_t* Ab  = (bf16_t*)(ws + 40 * MB);  // 8 MiB  [4096][1024] attn output

  const int nx = 4096 * 1024;
  const int nw = 1024 * 1024;
  f32_to_bf16<<<nx / 256, 256, 0, stream>>>(x,  xb,  nx);
  f32_to_bf16<<<nw / 256, 256, 0, stream>>>(Wq, wqb, nw);
  f32_to_bf16<<<nw / 256, 256, 0, stream>>>(Wk, wkb, nw);
  f32_to_bf16<<<nw / 256, 256, 0, stream>>>(Wv, wvb, nw);
  f32_to_bf16<<<nw / 256, 256, 0, stream>>>(Wo, wob, nw);

  qkv_gemm<<<dim3(128, 16, 3), 32, 0, stream>>>(xb, wqb, wkb, wvb, Qb, Kb, VTb);
  sparse_attn<<<dim3(256, 16), 32, 0, stream>>>(Qb, Kb, VTb, Ab);
  out_gemm<<<dim3(128, 16), 32, 0, stream>>>(Ab, wob, (float*)d_out);
}
